// HierarchicalContrast_79748952752297
// MI455X (gfx1250) — compile-verified, matrix operand order checked by reading, not run on hardware
//
#include <hip/hip_runtime.h>
#include <hip/hip_bf16.h>
#include <math.h>

typedef __attribute__((ext_vector_type(16))) _Float16 v16h;
typedef __attribute__((ext_vector_type(8)))  _Float16 v8h;
typedef __attribute__((ext_vector_type(4)))  _Float16 v4h;
typedef __attribute__((ext_vector_type(8)))  float    v8f;

#define BATCH 16
#define NNODE 4096
#define TTOK  512
#define PDIM  256
#define LY_LD 260   // f32 LDS stride for 16x256 staging (avoids 2-way conflicts)

// ---------------------------------------------------------------------------
// WMMA A/B fragment loaders for "NT" GEMM (dot of two row-major rows over K).
// CDNA5 16-bit A layout: lanes 0-15 hold K={k0..k0+7, k0+16..k0+23},
// lanes 16-31 hold K={k0+8..k0+15, k0+24..k0+31}. B (K x N) is the mirror
// pattern indexed by column, so the same loader serves both operands.
// ---------------------------------------------------------------------------
__device__ __forceinline__ v16h load_frag_nt(const _Float16* __restrict__ base,
                                             int r0, int ld, int k0,
                                             int lane, int rmax) {
  int row = r0 + (lane & 15);
  row = row < rmax ? row : (rmax - 1);          // clamp; writes are masked
  const _Float16* p = base + (size_t)row * ld + k0 + ((lane >> 4) << 3);
  v8h c0 = *reinterpret_cast<const v8h*>(p);      // K +0..7  (or +8..15)
  v8h c1 = *reinterpret_cast<const v8h*>(p + 16); // K +16..23 (or +24..31)
  v16h f;
#pragma unroll
  for (int i = 0; i < 8; ++i) { f[i] = c0[i]; f[i + 8] = c1[i]; }
  return f;
}

// Same fragment pattern, but sourced from an LDS slab (16 rows, stride ldp).
__device__ __forceinline__ v16h load_frag_lds(const _Float16* xs, int ldp,
                                              int k0, int lane) {
  const _Float16* p = xs + (lane & 15) * ldp + k0 + ((lane >> 4) << 3);
  v8h c0 = *reinterpret_cast<const v8h*>(p);
  v8h c1 = *reinterpret_cast<const v8h*>(p + 16);
  v16h f;
#pragma unroll
  for (int i = 0; i < 8; ++i) { f[i] = c0[i]; f[i + 8] = c1[i]; }
  return f;
}

// f32 -> f16 conversion, 4 elements/thread (all sizes are multiples of 4)
__global__ void cvt_f16_kernel(const float* __restrict__ x,
                               _Float16* __restrict__ y, int n) {
  int i = (blockIdx.x * blockDim.x + threadIdx.x) * 4;
  if (i + 3 < n) {
    float4 f = *reinterpret_cast<const float4*>(x + i);
    v4h h; h[0] = (_Float16)f.x; h[1] = (_Float16)f.y;
    h[2] = (_Float16)f.z; h[3] = (_Float16)f.w;
    *reinterpret_cast<v4h*>(y + i) = h;
  }
}

// Per-batch contiguous row ranges from sorted batch_idx, padded KT offsets
// (row length rounded to 8 halves for b128 loads), and num_ex.
__global__ void ranges_kernel(const int* __restrict__ bidx, int n,
                              int* __restrict__ rs, int* __restrict__ re,
                              int* __restrict__ kto, int* __restrict__ num_ex) {
  __shared__ int srs[BATCH], sre[BATCH];
  int b = threadIdx.x;
  if (b < BATCH) {
    int lo = 0, hi = n;
    while (lo < hi) { int m = (lo + hi) >> 1; if (bidx[m] < b) lo = m + 1; else hi = m; }
    srs[b] = lo;
    lo = 0; hi = n;
    while (lo < hi) { int m = (lo + hi) >> 1; if (bidx[m] < b + 1) lo = m + 1; else hi = m; }
    sre[b] = lo;
    rs[b] = srs[b]; re[b] = sre[b];
  }
  __syncthreads();
  if (b == 0) {
    int off = 0;
    for (int i = 0; i < BATCH; ++i) {
      kto[i] = off;
      int cnt = sre[i] - srs[i];
      int cntp = (cnt + 7) & ~7;
      off += TTOK * cntp;
    }
    int mx = bidx[n - 1] + 1;
    *num_ex = mx < BATCH ? mx : BATCH;
  }
}

// ---------------------------------------------------------------------------
// WMMA projection + bias + row-L2-normalize:  Y16 = normalize(X @ W^T + bias)
// X slab (16 rows x K) staged once into LDS (padded stride -> conflict-free
// ds_load_b128 A-fragments); wave w owns column tile w reading W from global.
// ---------------------------------------------------------------------------
__global__ void __launch_bounds__(512)
proj_norm_kernel(const _Float16* __restrict__ X, const _Float16* __restrict__ W,
                 const float* __restrict__ bias, _Float16* __restrict__ Y,
                 int M, int K) {
  extern __shared__ char smem[];
  const int ldp = K + 8;
  _Float16* xs = (_Float16*)smem;                       // [16][K+8]
  float*    ly = (float*)(smem + (size_t)16 * ldp * 2); // [16][LY_LD]
  int tid = threadIdx.x, wave = tid >> 5, lane = tid & 31;
  int m0 = blockIdx.x * 16;
  int n0 = wave * 16;

  // cooperative slab load: 16*K/8 b128 chunks
  int cpr = K >> 3;                       // chunks per row
  for (int c = tid; c < 16 * cpr; c += 512) {
    int r = c / cpr;
    int kk = (c - r * cpr) << 3;
    *reinterpret_cast<v8h*>(xs + r * ldp + kk) =
        *reinterpret_cast<const v8h*>(X + (size_t)(m0 + r) * K + kk);
  }
  __syncthreads();

  v8f acc = {};
  for (int k0 = 0; k0 < K; k0 += 32) {
    v16h a = load_frag_lds(xs, ldp, k0, lane);
    v16h b = load_frag_nt(W, n0, K, k0, lane, PDIM);
    acc = __builtin_amdgcn_wmma_f32_16x16x32_f16(false, a, false, b,
                                                 (short)0, acc, false, false);
  }
  int col = n0 + (lane & 15);
  float bv = bias[col];
  int rb = (lane >> 4) << 3;
#pragma unroll
  for (int k = 0; k < 8; ++k) ly[(rb + k) * LY_LD + col] = acc[k] + bv;
  __syncthreads();
  // wave w normalizes row w (256 elements, 8 per lane)
  float ss = 0.f;
#pragma unroll
  for (int j = 0; j < 8; ++j) { float x = ly[wave * LY_LD + lane + 32 * j]; ss += x * x; }
#pragma unroll
  for (int off = 16; off > 0; off >>= 1) ss += __shfl_xor(ss, off, 32);
  float scale = 1.0f / fmaxf(sqrtf(ss), 1e-12f);
#pragma unroll
  for (int j = 0; j < 8; ++j) {
    int c = lane + 32 * j;
    Y[(size_t)(m0 + wave) * PDIM + c] = (_Float16)(ly[wave * LY_LD + c] * scale);
  }
}

// ---------------------------------------------------------------------------
// Per-batch cost/kernel matrices via WMMA. B fragments (token tile) are
// hoisted into registers across the row-tile loop. Outputs f16:
//   Keff [N,512] row-major, Ceff [N,512], KT per-batch transposed with
//   row length padded to cntp=round8(cnt) (padding zero-filled).
// grid = (16 batches, 32 column tiles), 8 waves/block over row tiles.
// ---------------------------------------------------------------------------
__global__ void __launch_bounds__(256)
cost_kernel(const _Float16* __restrict__ nodesP, const _Float16* __restrict__ toksP,
            const int* __restrict__ rs_arr, const int* __restrict__ re_arr,
            const int* __restrict__ kto,
            _Float16* __restrict__ Keff, _Float16* __restrict__ Ceff,
            _Float16* __restrict__ KT) {
  int b = blockIdx.x;
  int t0 = blockIdx.y * 16;
  int wave = threadIdx.x >> 5;   // 0..7
  int lane = threadIdx.x & 31;
  int rs = rs_arr[b], re = re_arr[b], cnt = re - rs;
  if (cnt <= 0) return;
  int cntp = (cnt + 7) & ~7;
  int ntiles = (cnt + 15) >> 4;
  const _Float16* tokB = toksP + (size_t)b * TTOK * PDIM;
  _Float16* KTb = KT + (size_t)kto[b];

  v16h bfrag[8];
#pragma unroll
  for (int kk = 0; kk < 8; ++kk)
    bfrag[kk] = load_frag_nt(tokB, t0, PDIM, kk * 32, lane, TTOK);

  int col = t0 + (lane & 15);
  int rb = (lane >> 4) << 3;
  for (int tile = wave; tile < ntiles; tile += 8) {
    int r0 = rs + tile * 16;
    v8f acc = {};
#pragma unroll
    for (int kk = 0; kk < 8; ++kk) {
      v16h a = load_frag_nt(nodesP, r0, PDIM, kk * 32, lane, re);
      acc = __builtin_amdgcn_wmma_f32_16x16x32_f16(false, a, false, bfrag[kk],
                                                   (short)0, acc, false, false);
    }
#pragma unroll
    for (int k = 0; k < 8; ++k) {
      int m = r0 + rb + k;
      if (m < re) {
        float dot = acc[k];
        float kv = __expf(dot * 10.0f);              // exp(-C/OT_REG)
        Keff[(size_t)m * TTOK + col] = (_Float16)kv;
        Ceff[(size_t)m * TTOK + col] = (_Float16)(-dot);
        KTb[(size_t)col * cntp + (m - rs)] = (_Float16)kv;
      } else if (m - rs < cntp) {
        KTb[(size_t)col * cntp + (m - rs)] = (_Float16)0.f;  // pad for b128
      }
    }
  }
}

// ---------------------------------------------------------------------------
// Persistent per-batch Sinkhorn: 16 blocks x 32 waves, u/v in LDS, f16 K
// blocks stay L2-resident. All K/KT reads are b128 (8 halves/lane).
// 100 iterations, then sum(P.C).
// ---------------------------------------------------------------------------
__global__ void __launch_bounds__(1024)
sinkhorn_kernel(const _Float16* __restrict__ Keff, const _Float16* __restrict__ Ceff,
                const _Float16* __restrict__ KT,
                const int* __restrict__ rs_arr, const int* __restrict__ re_arr,
                const int* __restrict__ kto,
                float* __restrict__ per_batch) {
  __shared__ float u[NNODE];
  __shared__ float v[TTOK];
  __shared__ float partial[32];
  int b = blockIdx.x;
  int rs = rs_arr[b], re = re_arr[b], cnt = re - rs;
  int tid = threadIdx.x, wave = tid >> 5, lane = tid & 31;
  if (cnt <= 0) { if (tid == 0) per_batch[b] = 0.f; return; }
  int cntp = (cnt + 7) & ~7;
  float ainv = 1.0f / (float)cnt;
  const float bm = 1.0f / (float)TTOK;
  for (int j = tid; j < cntp; j += 1024) u[j] = (j < cnt) ? ainv : 0.f;
  for (int t = tid; t < TTOK; t += 1024) v[t] = bm;
  __syncthreads();
  const _Float16* Kb  = Keff + (size_t)rs * TTOK;
  const _Float16* KTb = KT + (size_t)kto[b];
  for (int it = 0; it < 100; ++it) {
    // v = bm / max(K^T u, eps)   (transposed copy, padded rows -> pure b128)
    for (int t = wave; t < TTOK; t += 32) {
      const _Float16* kt = KTb + (size_t)t * cntp;
      float s = 0.f;
      for (int jj = lane * 8; jj < cntp; jj += 256) {
        v8h k8 = *reinterpret_cast<const v8h*>(kt + jj);
        const float* up = &u[jj];
#pragma unroll
        for (int i = 0; i < 8; ++i) s += (float)k8[i] * up[i];
      }
#pragma unroll
      for (int off = 16; off > 0; off >>= 1) s += __shfl_xor(s, off, 32);
      if (lane == 0) v[t] = bm / fmaxf(s, 1e-16f);
    }
    __syncthreads();
    // u = a / max(K v, eps)
    for (int j = wave; j < cnt; j += 32) {
      const _Float16* kr = Kb + (size_t)j * TTOK;
      float s = 0.f;
#pragma unroll
      for (int c = 0; c < 2; ++c) {
        int t = lane * 8 + c * 256;
        v8h k8 = *reinterpret_cast<const v8h*>(kr + t);
#pragma unroll
        for (int i = 0; i < 8; ++i) s += (float)k8[i] * v[t + i];
      }
#pragma unroll
      for (int off = 16; off > 0; off >>= 1) s += __shfl_xor(s, off, 32);
      if (lane == 0) u[j] = ainv / fmaxf(s, 1e-16f);
    }
    __syncthreads();
  }
  // per-batch loss: sum_{n,t} u_n K_nt v_t C_nt
  const _Float16* Cb = Ceff + (size_t)rs * TTOK;
  float acc = 0.f;
  for (int j = wave; j < cnt; j += 32) {
    const _Float16* kr = Kb + (size_t)j * TTOK;
    const _Float16* cr = Cb + (size_t)j * TTOK;
    float s = 0.f;
#pragma unroll
    for (int c = 0; c < 2; ++c) {
      int t = lane * 8 + c * 256;
      v8h k8 = *reinterpret_cast<const v8h*>(kr + t);
      v8h c8 = *reinterpret_cast<const v8h*>(cr + t);
#pragma unroll
      for (int i = 0; i < 8; ++i) s += (float)k8[i] * v[t + i] * (float)c8[i];
    }
    acc += u[j] * s;
  }
#pragma unroll
  for (int off = 16; off > 0; off >>= 1) acc += __shfl_xor(acc, off, 32);
  if (lane == 0) partial[wave] = acc;
  __syncthreads();
  if (wave == 0) {
    float s = partial[lane];
#pragma unroll
    for (int off = 16; off > 0; off >>= 1) s += __shfl_xor(s, off, 32);
    if (lane == 0) per_batch[b] = s;
  }
}

// ---------------------------------------------------------------------------
// NT-Xent tail: one 16x16 WMMA tile for sim = s.q^T / TEMP, then dual lse.
// ---------------------------------------------------------------------------
__global__ void __launch_bounds__(32)
ntxent_kernel(const _Float16* __restrict__ s16, const _Float16* __restrict__ q16,
              float* __restrict__ g_out) {
  __shared__ float sim[16][17];
  int lane = threadIdx.x;
  v8f acc = {};
#pragma unroll
  for (int k0 = 0; k0 < PDIM; k0 += 32) {
    v16h a = load_frag_nt(s16, 0, PDIM, k0, lane, 16);
    v16h b = load_frag_nt(q16, 0, PDIM, k0, lane, 16);
    acc = __builtin_amdgcn_wmma_f32_16x16x32_f16(false, a, false, b,
                                                 (short)0, acc, false, false);
  }
  int col = lane & 15, rb = (lane >> 4) << 3;
#pragma unroll
  for (int k = 0; k < 8; ++k) sim[rb + k][col] = acc[k] * 10.0f;  // /TEMP
  __syncthreads();
  float contrib;
  if (lane < 16) {                    // row logsumexp
    int i = lane;
    float m = -1e30f;
    for (int j = 0; j < 16; ++j) m = fmaxf(m, sim[i][j]);
    float se = 0.f;
    for (int j = 0; j < 16; ++j) se += __expf(sim[i][j] - m);
    contrib = m + __logf(se) - sim[i][i];
  } else {                            // column logsumexp
    int j = lane - 16;
    float m = -1e30f;
    for (int i = 0; i < 16; ++i) m = fmaxf(m, sim[i][j]);
    float se = 0.f;
    for (int i = 0; i < 16; ++i) se += __expf(sim[i][j] - m);
    contrib = m + __logf(se) - sim[j][j];
  }
#pragma unroll
  for (int off = 16; off > 0; off >>= 1) contrib += __shfl_xor(contrib, off, 32);
  if (lane == 0) *g_out = contrib * (1.0f / 32.0f);  // (mean_i + mean_j)/2
}

__global__ void final_kernel(const float* __restrict__ g,
                             const float* __restrict__ per_batch,
                             const int* __restrict__ num_ex,
                             float* __restrict__ out) {
  if (threadIdx.x == 0) {
    float l = 0.f;
    for (int b = 0; b < BATCH; ++b) l += per_batch[b];
    int ne = *num_ex; if (ne < 1) ne = 1;
    l /= (float)ne;
    out[0] = *g; out[1] = l; out[2] = *g + l;
  }
}

// ---------------------------------------------------------------------------
extern "C" void kernel_launch(void* const* d_in, const int* in_sizes, int n_in,
                              void* d_out, int out_size, void* d_ws, size_t ws_size,
                              hipStream_t stream) {
  const float* sg   = (const float*)d_in[0];   // [16,256]
  const float* qg   = (const float*)d_in[1];   // [16,1280]
  const float* node = (const float*)d_in[2];   // [4096,256]
  const float* tok  = (const float*)d_in[3];   // [16,512,1280]
  const int*   bidx = (const int*)d_in[4];     // [4096]
  const float* Ws = (const float*)d_in[5];  const float* bs = (const float*)d_in[6];
  const float* Wq = (const float*)d_in[7];  const float* bq = (const float*)d_in[8];
  const float* Wn = (const float*)d_in[9];  const float* bn = (const float*)d_in[10];
  const float* Wt = (const float*)d_in[11]; const float* bt = (const float*)d_in[12];
  float* out = (float*)d_out;

  char* w = (char*)d_ws;
  size_t off = 0;
  auto carve = [&](size_t bytes) -> void* {
    void* p = w + off;
    off = (off + bytes + 255) & ~(size_t)255;
    return p;
  };
  _Float16* tok16   = (_Float16*)carve((size_t)8192 * 1280 * 2);
  _Float16* node16  = (_Float16*)carve((size_t)NNODE * PDIM * 2);
  _Float16* Wt16    = (_Float16*)carve((size_t)PDIM * 1280 * 2);
  _Float16* Wn16    = (_Float16*)carve((size_t)PDIM * PDIM * 2);
  _Float16* sg16    = (_Float16*)carve((size_t)16 * 256 * 2);
  _Float16* qg16    = (_Float16*)carve((size_t)16 * 1280 * 2);
  _Float16* Ws16    = (_Float16*)carve((size_t)PDIM * 256 * 2);
  _Float16* Wq16    = (_Float16*)carve((size_t)PDIM * 1280 * 2);
  _Float16* toksP   = (_Float16*)carve((size_t)8192 * PDIM * 2);
  _Float16* nodesP  = (_Float16*)carve((size_t)NNODE * PDIM * 2);
  _Float16* s16n    = (_Float16*)carve((size_t)16 * PDIM * 2);
  _Float16* q16n    = (_Float16*)carve((size_t)16 * PDIM * 2);
  _Float16* Keff    = (_Float16*)carve((size_t)NNODE * TTOK * 2);
  _Float16* Ceff    = (_Float16*)carve((size_t)NNODE * TTOK * 2);
  _Float16* KT      = (_Float16*)carve((size_t)(NNODE + 128) * TTOK * 2); // padded
  float*    per_b   = (float*)carve(BATCH * 4);
  float*    g_ptr   = (float*)carve(4);
  int*      rs      = (int*)carve(BATCH * 4);
  int*      re      = (int*)carve(BATCH * 4);
  int*      kto     = (int*)carve(BATCH * 4);
  int*      num_ex  = (int*)carve(4);

  ranges_kernel<<<1, 32, 0, stream>>>(bidx, NNODE, rs, re, kto, num_ex);

  auto cvt = [&](const float* src, _Float16* dst, int n) {
    cvt_f16_kernel<<<(n / 4 + 255) / 256, 256, 0, stream>>>(src, dst, n);
  };
  cvt(tok,  tok16,  8192 * 1280);
  cvt(node, node16, NNODE * PDIM);
  cvt(Wt,   Wt16,   PDIM * 1280);
  cvt(Wn,   Wn16,   PDIM * PDIM);
  cvt(sg,   sg16,   16 * 256);
  cvt(qg,   qg16,   16 * 1280);
  cvt(Ws,   Ws16,   PDIM * 256);
  cvt(Wq,   Wq16,   PDIM * 1280);

  // Projections + normalize (WMMA f16 -> f32), X slab staged in LDS
  auto smem = [](int K) { return (size_t)16 * (K + 8) * 2 + (size_t)16 * LY_LD * 4; };
  proj_norm_kernel<<<8192 / 16, 512, smem(1280), stream>>>(tok16,  Wt16, bt, toksP,  8192, 1280);
  proj_norm_kernel<<<NNODE / 16, 512, smem(256), stream>>>(node16, Wn16, bn, nodesP, NNODE, PDIM);
  proj_norm_kernel<<<1, 512, smem(256),  stream>>>(sg16, Ws16, bs, s16n, 16, 256);
  proj_norm_kernel<<<1, 512, smem(1280), stream>>>(qg16, Wq16, bq, q16n, 16, 1280);

  // Global NT-Xent
  ntxent_kernel<<<1, 32, 0, stream>>>(s16n, q16n, g_ptr);

  // Per-batch cost/K matrices (block-compacted [4096,512], f16)
  cost_kernel<<<dim3(BATCH, TTOK / 16), 256, 0, stream>>>(nodesP, toksP, rs, re, kto,
                                                          Keff, Ceff, KT);

  // Persistent per-batch Sinkhorn (100 iters, L2-resident f16 K, b128 loads)
  sinkhorn_kernel<<<BATCH, 1024, 0, stream>>>(Keff, Ceff, KT, rs, re, kto, per_b);

  final_kernel<<<1, 32, 0, stream>>>(g_ptr, per_b, num_ex, out);
}
